// GCNModel_13108240187969
// MI455X (gfx1250) — matrix operand order repaired and checked by reading, not compiled, not run on hardware
//
#include <hip/hip_runtime.h>

// ---------------- problem constants (match reference) ----------------
#define WS_B   8
#define WS_N   1536
#define WS_F   256
#define WS_E   24576
#define WS_BN  (WS_B * WS_N)        // 12288 nodes total
#define WS_NN  (WS_N * WS_N)        // 2359296 per-branch adjacency

// ---------------- WMMA tiling ----------------
#define TM 128                      // block tile M
#define TN 64                       // block tile N
#define TK 32                       // K step (bf16 WMMA K)
#define ASTR 34                     // padded LDS row stride (bf16 elems)

typedef __attribute__((ext_vector_type(16))) __bf16 v16bf;
typedef __attribute__((ext_vector_type(8)))  float  v8f;

union BFV { v16bf v; unsigned short u[16]; };

// fp32 -> bf16, round-to-nearest-even (bit manipulation; no __bf16 arithmetic)
__device__ __forceinline__ unsigned short f2bf(float f) {
  unsigned u = __float_as_uint(f);
  return (unsigned short)((u + 0x7FFFu + ((u >> 16) & 1u)) >> 16);
}

// A fragment (16x32 bf16): lane<16 -> M=lane, K in {kb..kb+7} U {16+kb..16+kb+7}, kb = lane>=16 ? 8 : 0
__device__ __forceinline__ v16bf frag_a(const unsigned short* As, int mbase, int lane) {
  BFV r;
  const unsigned short* row = As + (mbase + (lane & 15)) * ASTR;
  const int kb = (lane & 16) ? 8 : 0;
#pragma unroll
  for (int p = 0; p < 8; ++p) r.u[p] = row[kb + p];
#pragma unroll
  for (int p = 0; p < 8; ++p) r.u[8 + p] = row[16 + kb + p];
  return r.v;
}

// B fragment (32x16 bf16), Bs stored [n][k] (k contiguous): K in {kb..kb+7} U {16+kb..16+kb+7}, kb = lane>=16 ? 8 : 0
__device__ __forceinline__ v16bf frag_b(const unsigned short* Bs, int nbase, int lane) {
  BFV r;
  const unsigned short* row = Bs + (nbase + (lane & 15)) * ASTR;
  const int kb = (lane & 16) ? 8 : 0;
#pragma unroll
  for (int p = 0; p < 8; ++p) r.u[p] = row[kb + p];
#pragma unroll
  for (int p = 0; p < 8; ++p) r.u[8 + p] = row[16 + kb + p];
  return r.v;
}

__device__ __forceinline__ v8f wmma_bf16(v16bf a, v16bf b, v8f c) {
  return __builtin_amdgcn_wmma_f32_16x16x32_bf16(false, a, false, b, (short)0, c, false, false);
}

// Per-wave 32x32 tile: 2 A frags x 2 B frags -> 4 WMMA per K step.
#define WAVE_DECOMP()                                                  \
  const int tid = threadIdx.x, lane = tid & 31, wave = tid >> 5;       \
  const int m_off = (wave >> 1) * 32, n_off = (wave & 1) * 32;         \
  v8f acc00 = {}; v8f acc01 = {}; v8f acc10 = {}; v8f acc11 = {};

#define WAVE_MMA_STEP()                                                \
  {                                                                    \
    v16bf a0 = frag_a(As, m_off, lane);                                \
    v16bf a1 = frag_a(As, m_off + 16, lane);                           \
    v16bf b0 = frag_b(Bs, n_off, lane);                                \
    v16bf b1 = frag_b(Bs, n_off + 16, lane);                           \
    acc00 = wmma_bf16(a0, b0, acc00);                                  \
    acc01 = wmma_bf16(a0, b1, acc01);                                  \
    acc10 = wmma_bf16(a1, b0, acc10);                                  \
    acc11 = wmma_bf16(a1, b1, acc11);                                  \
  }

// ====================================================================
// Generic feature GEMM: C[M,No] = A[M,K] @ W[K,No]   (fp32 in/out, bf16 WMMA)
// ====================================================================
__global__ void gemm_xw_kernel(const float* __restrict__ A, const float* __restrict__ W,
                               float* __restrict__ C, int M, int K, int No) {
  __shared__ unsigned short As[TM * ASTR];
  __shared__ unsigned short Bs[TN * ASTR];
  const int mt = blockIdx.x * TM, nt = blockIdx.y * TN;
  WAVE_DECOMP();
  for (int k0 = 0; k0 < K; k0 += TK) {
    for (int idx = tid; idx < TM * TK; idx += 256) {
      int m = idx >> 5, k = idx & 31;
      As[m * ASTR + k] = f2bf(A[(size_t)(mt + m) * K + k0 + k]);
    }
    for (int idx = tid; idx < TN * TK; idx += 256) {
      int k = idx >> 6, n = idx & 63;
      Bs[n * ASTR + k] = f2bf(W[(size_t)(k0 + k) * No + nt + n]);
    }
    if (k0 + TK < K) {  // prefetch next K tile (global_prefetch_b8, no counter cost)
      __builtin_prefetch(&A[(size_t)(mt + (tid >> 1)) * K + k0 + TK + ((tid & 1) << 4)], 0, 0);
      __builtin_prefetch(&W[(size_t)(k0 + TK + (tid >> 3)) * No + nt + ((tid & 7) << 3)], 0, 0);
    }
    __syncthreads();
    WAVE_MMA_STEP();
    __syncthreads();
  }
  const int em = (lane & 16) ? 8 : 0, en = lane & 15;
#pragma unroll
  for (int r = 0; r < 8; ++r) {
    int gm0 = mt + m_off + r + em, gm1 = gm0 + 16;
    int gn0 = nt + n_off + en,     gn1 = gn0 + 16;
    C[(size_t)gm0 * No + gn0] = acc00[r];
    C[(size_t)gm0 * No + gn1] = acc01[r];
    C[(size_t)gm1 * No + gn0] = acc10[r];
    C[(size_t)gm1 * No + gn1] = acc11[r];
  }
}

// ====================================================================
// adj[b,i,j] = relu(X[b,i,:]·Wa[j,:] + ba[j]);  scal[b] += sum(adj^2)
// ====================================================================
__global__ void adj_gemm_kernel(const float* __restrict__ X, const float* __restrict__ Wa,
                                const float* __restrict__ ba, float* __restrict__ adj,
                                float* __restrict__ scal) {
  __shared__ unsigned short As[TM * ASTR];
  __shared__ unsigned short Bs[TN * ASTR];
  __shared__ float red[256];
  const int b = blockIdx.z;
  const int it = blockIdx.x * TM, jt = blockIdx.y * TN;
  const float* Xb = X + (size_t)b * WS_N * WS_F;
  WAVE_DECOMP();
  for (int k0 = 0; k0 < WS_F; k0 += TK) {
    for (int idx = tid; idx < TM * TK; idx += 256) {
      int m = idx >> 5, k = idx & 31;
      As[m * ASTR + k] = f2bf(Xb[(size_t)(it + m) * WS_F + k0 + k]);
    }
    for (int idx = tid; idx < TN * TK; idx += 256) {
      int n = idx >> 5, k = idx & 31;
      Bs[n * ASTR + k] = f2bf(Wa[(size_t)(jt + n) * WS_F + k0 + k]);
    }
    if (k0 + TK < WS_F) {
      __builtin_prefetch(&Xb[(size_t)(it + (tid >> 1)) * WS_F + k0 + TK + ((tid & 1) << 4)], 0, 0);
      __builtin_prefetch(&Wa[(size_t)(jt + (tid >> 2)) * WS_F + k0 + TK + ((tid & 3) << 3)], 0, 0);
    }
    __syncthreads();
    WAVE_MMA_STEP();
    __syncthreads();
  }
  const int em = (lane & 16) ? 8 : 0, en = lane & 15;
  float ss = 0.f;
  float* adjb = adj + (size_t)b * WS_NN;
#pragma unroll
  for (int r = 0; r < 8; ++r) {
    int gi0 = it + m_off + r + em, gi1 = gi0 + 16;
    int gj0 = jt + n_off + en,     gj1 = gj0 + 16;
    float v00 = acc00[r] + ba[gj0]; v00 = v00 > 0.f ? v00 : 0.f;
    float v01 = acc01[r] + ba[gj1]; v01 = v01 > 0.f ? v01 : 0.f;
    float v10 = acc10[r] + ba[gj0]; v10 = v10 > 0.f ? v10 : 0.f;
    float v11 = acc11[r] + ba[gj1]; v11 = v11 > 0.f ? v11 : 0.f;
    adjb[(size_t)gi0 * WS_N + gj0] = v00;
    adjb[(size_t)gi0 * WS_N + gj1] = v01;
    adjb[(size_t)gi1 * WS_N + gj0] = v10;
    adjb[(size_t)gi1 * WS_N + gj1] = v11;
    ss += v00 * v00 + v01 * v01 + v10 * v10 + v11 * v11;
  }
  red[tid] = ss; __syncthreads();
  for (int s = 128; s > 0; s >>= 1) { if (tid < s) red[tid] += red[tid + s]; __syncthreads(); }
  if (tid == 0) atomicAdd(&scal[b], red[0]);
}

// ====================================================================
// sim = xn·xnᵀ via WMMA; a = adj*invfro; loss += (a-sim)^2; gmax = max(a);
// writes normalized a back into adj.
// scal layout: [0..7] fro2, [8..15] invfro, [16] gmax (float bits), [17] thr, [18] loss
// ====================================================================
__global__ void sim_loss_kernel(const float* __restrict__ X, const float* __restrict__ invn,
                                float* __restrict__ adj, float* __restrict__ scal) {
  __shared__ unsigned short As[TM * ASTR];
  __shared__ unsigned short Bs[TN * ASTR];
  __shared__ float redl[256];
  __shared__ float redm[256];
  const int b = blockIdx.z;
  const int it = blockIdx.x * TM, jt = blockIdx.y * TN;
  WAVE_DECOMP();
  for (int k0 = 0; k0 < WS_F; k0 += TK) {
    for (int idx = tid; idx < TM * TK; idx += 256) {
      int m = idx >> 5, k = idx & 31;
      int gn = b * WS_N + it + m;
      As[m * ASTR + k] = f2bf(X[(size_t)gn * WS_F + k0 + k] * invn[gn]);
    }
    for (int idx = tid; idx < TN * TK; idx += 256) {
      int n = idx >> 5, k = idx & 31;
      int gn = b * WS_N + jt + n;
      Bs[n * ASTR + k] = f2bf(X[(size_t)gn * WS_F + k0 + k] * invn[gn]);
    }
    if (k0 + TK < WS_F) {
      __builtin_prefetch(&X[(size_t)(b * WS_N + it + (tid >> 1)) * WS_F + k0 + TK + ((tid & 1) << 4)], 0, 0);
      __builtin_prefetch(&X[(size_t)(b * WS_N + jt + (tid >> 2)) * WS_F + k0 + TK + ((tid & 3) << 3)], 0, 0);
    }
    __syncthreads();
    WAVE_MMA_STEP();
    __syncthreads();
  }
  const int em = (lane & 16) ? 8 : 0, en = lane & 15;
  const float invf = scal[8 + b];
  float* adjb = adj + (size_t)b * WS_NN;
  float lsum = 0.f, lmax = 0.f;
#pragma unroll
  for (int r = 0; r < 8; ++r) {
    int gi0 = it + m_off + r + em, gi1 = gi0 + 16;
    int gj0 = jt + n_off + en,     gj1 = gj0 + 16;
    size_t o00 = (size_t)gi0 * WS_N + gj0, o01 = (size_t)gi0 * WS_N + gj1;
    size_t o10 = (size_t)gi1 * WS_N + gj0, o11 = (size_t)gi1 * WS_N + gj1;
    float a00 = adjb[o00] * invf; float d00 = a00 - acc00[r];
    float a01 = adjb[o01] * invf; float d01 = a01 - acc01[r];
    float a10 = adjb[o10] * invf; float d10 = a10 - acc10[r];
    float a11 = adjb[o11] * invf; float d11 = a11 - acc11[r];
    lsum += d00 * d00 + d01 * d01 + d10 * d10 + d11 * d11;
    lmax = fmaxf(lmax, fmaxf(fmaxf(a00, a01), fmaxf(a10, a11)));
    adjb[o00] = a00; adjb[o01] = a01; adjb[o10] = a10; adjb[o11] = a11;
  }
  redl[tid] = lsum; redm[tid] = lmax; __syncthreads();
  for (int s = 128; s > 0; s >>= 1) {
    if (tid < s) { redl[tid] += redl[tid + s]; redm[tid] = fmaxf(redm[tid], redm[tid + s]); }
    __syncthreads();
  }
  if (tid == 0) {
    atomicAdd(&scal[18], redl[0]);
    atomicMax((int*)scal + 16, __float_as_int(redm[0]));  // values >= 0
  }
}

// ====================================================================
// out[b,j,f] = relu( sum_i S[b,i,j]*xw[b,i,f] + bias[f] )
// S built on-the-fly: mask = (adjn >= thr) || (i==j);  S = mask*dinv[i]*dinv[j]
// ====================================================================
__global__ void agg_gemm_kernel(const float* __restrict__ adj, const float* __restrict__ scal,
                                const float* __restrict__ dinv, const float* __restrict__ xw,
                                const float* __restrict__ bias, float* __restrict__ out) {
  __shared__ unsigned short As[TM * ASTR];
  __shared__ unsigned short Bs[TN * ASTR];
  const int b = blockIdx.z;
  const int jt = blockIdx.x * TM;   // output node tile (M dim = j)
  const int ft = blockIdx.y * TN;   // feature tile   (N dim = f)
  const float thr = scal[17];
  const float* adjb = adj + (size_t)b * WS_NN;
  const float* dvb  = dinv + b * WS_N;
  const float* xwb  = xw + (size_t)b * WS_N * WS_F;
  WAVE_DECOMP();
  for (int k0 = 0; k0 < WS_N; k0 += TK) {       // K dim = source node i
    for (int idx = tid; idx < TM * TK; idx += 256) {
      int k = idx >> 7, j = idx & 127;
      int gi = k0 + k, gj = jt + j;
      float v = adjb[(size_t)gi * WS_N + gj];
      unsigned short o = 0;
      if (v >= thr || gi == gj) o = f2bf(dvb[gi] * dvb[gj]);
      As[j * ASTR + k] = o;
    }
    for (int idx = tid; idx < TN * TK; idx += 256) {
      int k = idx >> 6, n = idx & 63;
      Bs[n * ASTR + k] = f2bf(xwb[(size_t)(k0 + k) * WS_F + ft + n]);
    }
    if (k0 + TK < WS_N) {
      __builtin_prefetch(&adjb[(size_t)(k0 + TK + (tid >> 3)) * WS_N + jt + ((tid & 7) << 4)], 0, 0);
      __builtin_prefetch(&xwb[(size_t)(k0 + TK + (tid >> 3)) * WS_F + ft + ((tid & 7) << 3)], 0, 0);
    }
    __syncthreads();
    WAVE_MMA_STEP();
    __syncthreads();
  }
  const int em = (lane & 16) ? 8 : 0, en = lane & 15;
#pragma unroll
  for (int r = 0; r < 8; ++r) {
    int gj0 = jt + m_off + r + em, gj1 = gj0 + 16;
    int gf0 = ft + n_off + en,     gf1 = gf0 + 16;
    float v00 = acc00[r] + bias[gf0]; v00 = v00 > 0.f ? v00 : 0.f;
    float v01 = acc01[r] + bias[gf1]; v01 = v01 > 0.f ? v01 : 0.f;
    float v10 = acc10[r] + bias[gf0]; v10 = v10 > 0.f ? v10 : 0.f;
    float v11 = acc11[r] + bias[gf1]; v11 = v11 > 0.f ? v11 : 0.f;
    out[(size_t)(b * WS_N + gj0) * WS_F + gf0] = v00;
    out[(size_t)(b * WS_N + gj0) * WS_F + gf1] = v01;
    out[(size_t)(b * WS_N + gj1) * WS_F + gf0] = v10;
    out[(size_t)(b * WS_N + gj1) * WS_F + gf1] = v11;
  }
}

// ====================================================================
// small helper kernels
// ====================================================================
__global__ void zero_kernel(float* p, int n) {
  int i = blockIdx.x * 256 + threadIdx.x;
  if (i < n) p[i] = 0.f;
}

__global__ void edge_deg_kernel(const int* __restrict__ se, int nE, int nBatch, int nOff,
                                float* __restrict__ dege, float* __restrict__ hself) {
  int g = blockIdx.x * 256 + threadIdx.x;
  if (g >= nE * nBatch) return;
  int b = g / nE, e = g - b * nE;
  int row = se[e] + b * nOff;
  int col = se[nE + e] + b * nOff;
  atomicAdd(&dege[col], 1.0f);
  if (row == col) atomicAdd(&hself[col], 1.0f);
}

__global__ void node_norm_kernel(const float* __restrict__ dege, const float* __restrict__ hself,
                                 float* __restrict__ dinv, float* __restrict__ selfc, int n) {
  int i = blockIdx.x * 256 + threadIdx.x;
  if (i >= n) return;
  float add_self = (hself[i] > 0.f) ? 0.f : 1.f;
  float deg = dege[i] + add_self;
  float di = rsqrtf(deg);
  dinv[i] = di;
  selfc[i] = add_self * di * di;
}

__global__ void init_out_kernel(const float* __restrict__ xw, const float* __restrict__ selfc,
                                float* __restrict__ out, int total) {
  int t = blockIdx.x * 256 + threadIdx.x;
  if (t < total) out[t] = selfc[t >> 8] * xw[t];   // F = 256
}

__global__ void edge_msg_kernel(const int* __restrict__ se, int nE, int nOff,
                                const float* __restrict__ dinv, const float* __restrict__ xw,
                                float* __restrict__ out) {
  int g = blockIdx.x;
  int b = g / nE, e = g - b * nE;
  int row = se[e] + b * nOff;
  int col = se[nE + e] + b * nOff;
  float c = dinv[row] * dinv[col];
  int f = threadIdx.x;
  atomicAdd(&out[(size_t)col * WS_F + f], xw[(size_t)row * WS_F + f] * c);
}

__global__ void bias_act_kernel(float* __restrict__ buf, const float* __restrict__ bias,
                                int do_relu, int total) {
  int t = blockIdx.x * 256 + threadIdx.x;
  if (t >= total) return;
  float v = buf[t] + bias[t & (WS_F - 1)];
  if (do_relu) v = v > 0.f ? v : 0.f;
  buf[t] = v;
}

__global__ void invn_kernel(const float* __restrict__ X, float* __restrict__ invn) {
  __shared__ float red[256];
  int n = blockIdx.x, t = threadIdx.x;
  float v = X[(size_t)n * WS_F + t];
  red[t] = v * v; __syncthreads();
  for (int s = 128; s > 0; s >>= 1) { if (t < s) red[t] += red[t + s]; __syncthreads(); }
  if (t == 0) invn[n] = rsqrtf(red[0]);
}

__global__ void fro_fin_kernel(float* scal) {
  int b = threadIdx.x;
  if (b < WS_B) scal[8 + b] = rsqrtf(scal[b]);    // 1/fro = rsqrt(sum sq)
}

__global__ void thr_kernel(float* scal) {
  if (threadIdx.x == 0) scal[17] = 0.5f * __int_as_float(((int*)scal)[16]);
}

__global__ void col_deg_kernel(const float* __restrict__ adj, const float* __restrict__ scal,
                               float* __restrict__ deg) {
  int b = blockIdx.z;
  int j = blockIdx.x * 256 + threadIdx.x;
  int i0 = blockIdx.y * 256;
  float thr = scal[17];
  const float* adjb = adj + (size_t)b * WS_NN;
  float s = 0.f;
  for (int i = i0; i < i0 + 256; ++i) {
    float v = adjb[(size_t)i * WS_N + j];
    s += ((v >= thr) || (i == j)) ? 1.f : 0.f;
  }
  atomicAdd(&deg[b * WS_N + j], s);
}

__global__ void dinv_kernel(const float* __restrict__ deg, float* __restrict__ dinv, int n) {
  int i = blockIdx.x * 256 + threadIdx.x;
  if (i < n) dinv[i] = rsqrtf(deg[i]);
}

__global__ void readout_kernel(const float* __restrict__ X, const float* __restrict__ Wr,
                               const float* __restrict__ br, float* __restrict__ out) {
  __shared__ float r0[256];
  __shared__ float r1[256];
  int n = blockIdx.x, t = threadIdx.x;
  float v = X[(size_t)n * WS_F + t];
  r0[t] = v * Wr[t];
  r1[t] = v * Wr[WS_F + t];
  __syncthreads();
  for (int s = 128; s > 0; s >>= 1) {
    if (t < s) { r0[t] += r0[t + s]; r1[t] += r1[t + s]; }
    __syncthreads();
  }
  if (t == 0) {
    out[(size_t)n * 2 + 0] = r0[0] + br[0];
    out[(size_t)n * 2 + 1] = r1[0] + br[1];
  }
}

__global__ void loss_out_kernel(const float* __restrict__ scal, float* __restrict__ out) {
  if (threadIdx.x == 0) out[0] = scal[18] / (float)WS_B;  // mean over batch of per-batch sums
}

// ====================================================================
// launcher
// ====================================================================
extern "C" void kernel_launch(void* const* d_in, const int* in_sizes, int n_in,
                              void* d_out, int out_size, void* d_ws, size_t ws_size,
                              hipStream_t stream) {
  (void)in_sizes; (void)n_in; (void)out_size; (void)ws_size;
  const float* x_in = (const float*)d_in[0];
  const float* W1 = (const float*)d_in[1];  const float* b1 = (const float*)d_in[2];
  const float* W2 = (const float*)d_in[3];  const float* b2 = (const float*)d_in[4];
  const float* W3 = (const float*)d_in[5];  const float* b3 = (const float*)d_in[6];
  const float* W4 = (const float*)d_in[7];  const float* b4 = (const float*)d_in[8];
  const float* Wa = (const float*)d_in[9];  const float* ba = (const float*)d_in[10];
  const float* Wr = (const float*)d_in[11]; const float* br = (const float*)d_in[12];
  const int*   se = (const int*)d_in[13];   // [2, E]
  float* out = (float*)d_out;

  // ---- workspace carve (fp32 elements): ~114 MB total ----
  float* ws    = (float*)d_ws;
  float* xw    = ws;                                   // BN*F
  float* bufA  = xw   + (size_t)WS_BN * WS_F;          // BN*F
  float* bufB  = bufA + (size_t)WS_BN * WS_F;          // BN*F
  float* adj   = bufB + (size_t)WS_BN * WS_F;          // B*N*N
  float* dege  = adj  + (size_t)WS_B * WS_NN;          // BN
  float* hself = dege  + WS_BN;                        // BN
  float* dinv  = hself + WS_BN;                        // BN
  float* selfc = dinv  + WS_BN;                        // BN
  float* invn  = selfc + WS_BN;                        // BN
  float* deg2  = invn  + WS_BN;                        // BN
  float* scal  = deg2  + WS_BN;                        // 64 scalars

  const dim3 gFeat(WS_BN / TM, WS_F / TN);             // (96, 4)
  const dim3 gAdj(WS_N / TM, WS_N / TN, WS_B);         // (12, 24, 8)
  const dim3 gAgg(WS_N / TM, WS_F / TN, WS_B);         // (12, 4, 8)
  const int  nBNF = WS_BN * WS_F;

  zero_kernel<<<1, 256, 0, stream>>>(scal, 64);

  // ================= gcn1: sparse, tiled edges (B*E) =================
  zero_kernel<<<(WS_BN + 255) / 256, 256, 0, stream>>>(dege, WS_BN);
  zero_kernel<<<(WS_BN + 255) / 256, 256, 0, stream>>>(hself, WS_BN);
  gemm_xw_kernel<<<gFeat, 256, 0, stream>>>(x_in, W1, xw, WS_BN, WS_F, WS_F);
  edge_deg_kernel<<<(WS_B * WS_E + 255) / 256, 256, 0, stream>>>(se, WS_E, WS_B, WS_N, dege, hself);
  node_norm_kernel<<<(WS_BN + 255) / 256, 256, 0, stream>>>(dege, hself, dinv, selfc, WS_BN);
  init_out_kernel<<<(nBNF + 255) / 256, 256, 0, stream>>>(xw, selfc, bufA, nBNF);
  edge_msg_kernel<<<WS_B * WS_E, 256, 0, stream>>>(se, WS_E, WS_N, dinv, xw, bufA);
  bias_act_kernel<<<(nBNF + 255) / 256, 256, 0, stream>>>(bufA, b1, 1, nBNF);

  // ================= adj_net 1 (on bufA) =================
  invn_kernel<<<WS_BN, 256, 0, stream>>>(bufA, invn);
  adj_gemm_kernel<<<gAdj, 256, 0, stream>>>(bufA, Wa, ba, adj, scal);
  fro_fin_kernel<<<1, 32, 0, stream>>>(scal);
  sim_loss_kernel<<<gAdj, 256, 0, stream>>>(bufA, invn, adj, scal);
  thr_kernel<<<1, 32, 0, stream>>>(scal);

  // ================= gcn2: dense aggregation with mask1 =================
  zero_kernel<<<(WS_BN + 255) / 256, 256, 0, stream>>>(deg2, WS_BN);
  col_deg_kernel<<<dim3(WS_N / 256, WS_N / 256, WS_B), 256, 0, stream>>>(adj, scal, deg2);
  dinv_kernel<<<(WS_BN + 255) / 256, 256, 0, stream>>>(deg2, dinv, WS_BN);
  gemm_xw_kernel<<<gFeat, 256, 0, stream>>>(bufA, W2, xw, WS_BN, WS_F, WS_F);
  agg_gemm_kernel<<<gAgg, 256, 0, stream>>>(adj, scal, dinv, xw, b2, bufB);

  // ================= adj_net 2 (on bufB) =================
  zero_kernel<<<1, 256, 0, stream>>>(scal, 17);        // reset fro2[0..7] + gmax; keep loss
  invn_kernel<<<WS_BN, 256, 0, stream>>>(bufB, invn);
  adj_gemm_kernel<<<gAdj, 256, 0, stream>>>(bufB, Wa, ba, adj, scal);
  fro_fin_kernel<<<1, 32, 0, stream>>>(scal);
  sim_loss_kernel<<<gAdj, 256, 0, stream>>>(bufB, invn, adj, scal);
  thr_kernel<<<1, 32, 0, stream>>>(scal);

  // ================= gcn3: dense aggregation with mask2 =================
  zero_kernel<<<(WS_BN + 255) / 256, 256, 0, stream>>>(deg2, WS_BN);
  col_deg_kernel<<<dim3(WS_N / 256, WS_N / 256, WS_B), 256, 0, stream>>>(adj, scal, deg2);
  dinv_kernel<<<(WS_BN + 255) / 256, 256, 0, stream>>>(deg2, dinv, WS_BN);
  gemm_xw_kernel<<<gFeat, 256, 0, stream>>>(bufB, W3, xw, WS_BN, WS_F, WS_F);
  agg_gemm_kernel<<<gAgg, 256, 0, stream>>>(adj, scal, dinv, xw, b3, bufA);

  // ================= gcn4: sparse, UN-tiled edges on [B*N] nodes =================
  zero_kernel<<<(WS_BN + 255) / 256, 256, 0, stream>>>(dege, WS_BN);
  zero_kernel<<<(WS_BN + 255) / 256, 256, 0, stream>>>(hself, WS_BN);
  edge_deg_kernel<<<(WS_E + 255) / 256, 256, 0, stream>>>(se, WS_E, 1, 0, dege, hself);
  node_norm_kernel<<<(WS_BN + 255) / 256, 256, 0, stream>>>(dege, hself, dinv, selfc, WS_BN);
  gemm_xw_kernel<<<gFeat, 256, 0, stream>>>(bufA, W4, xw, WS_BN, WS_F, WS_F);
  init_out_kernel<<<(nBNF + 255) / 256, 256, 0, stream>>>(xw, selfc, bufB, nBNF);
  edge_msg_kernel<<<WS_E, 256, 0, stream>>>(se, WS_E, 0, dinv, xw, bufB);
  bias_act_kernel<<<(nBNF + 255) / 256, 256, 0, stream>>>(bufB, b4, 0, nBNF);  // no relu

  // ================= readout + loss =================
  readout_kernel<<<WS_BN, 256, 0, stream>>>(bufB, Wr, br, out);
  loss_out_kernel<<<1, 32, 0, stream>>>(scal, out + (size_t)WS_BN * 2);
}
